// MHAWithRelativeEmbeddings_57123065036934
// MI455X (gfx1250) — matrix operand order repaired and checked by reading, not compiled, not run on hardware
//
#include <hip/hip_runtime.h>
#include <hip/hip_bf16.h>

typedef __attribute__((ext_vector_type(16))) _Float16 v16h;
typedef __attribute__((ext_vector_type(8)))  _Float16 v8h;
typedef __attribute__((ext_vector_type(8)))  float    v8f;

#define T_SEQ 2048
#define DH    64
#define HEADS 16
#define DMODEL 1024
#define KDIM  1024
#define NCOLS 3072

// logits scale 0.5/sqrt(64)=0.0625, folded with log2(e) so softmax uses exp2 (v_exp_f32)
#define QSCALE     (0.0625f * 1.44269504088896f)
#define MASK_SCALE (-10000.0f * 1.44269504088896f)

__device__ __forceinline__ v8f wmma_f16(v16h a, v16h b, v8f c) {
  return __builtin_amdgcn_wmma_f32_16x16x32_f16(false, a, false, b, (short)0, c, false, false);
}

// A-fragment: lanes 0-15 hold K {0..15}, lanes 16-31 hold K {16..31}
__device__ __forceinline__ v16h ld_a(const _Float16* base, int lo, int hiOff) {
  union { v16h v; v8h h[2]; } u;
  u.h[0] = *(const v8h*)(base + lo);
  u.h[1] = *(const v8h*)(base + hiOff);
  return u.v;
}
// B-fragment from row-major [N][K] source: 16 contiguous K values per lane
__device__ __forceinline__ v16h ld_b(const _Float16* p) {
  union { v16h v; v8h h[2]; } u;
  u.h[0] = *(const v8h*)(p);
  u.h[1] = *(const v8h*)(p + 8);
  return u.v;
}

__device__ __forceinline__ float half_reduce_max(float v) {
  v = fmaxf(v, __shfl_xor(v, 1));
  v = fmaxf(v, __shfl_xor(v, 2));
  v = fmaxf(v, __shfl_xor(v, 4));
  v = fmaxf(v, __shfl_xor(v, 8));
  return v;
}
__device__ __forceinline__ float half_reduce_sum(float v) {
  v += __shfl_xor(v, 1);
  v += __shfl_xor(v, 2);
  v += __shfl_xor(v, 4);
  v += __shfl_xor(v, 8);
  return v;
}

// G = q_tile (16x64) x R[p0 .. p0+15]^T  -> 16x16 scores against 16 relative positions
__device__ __forceinline__ v8f rel_g(const _Float16* __restrict__ Rb, int p0,
                                     v16h qa0, v16h qa1, int l15, int hi) {
  const _Float16* row = Rb + (size_t)(p0 + l15) * DH + hi * 16;
  v8f g = {};
  g = wmma_f16(qa0, ld_b(row), g);
  g = wmma_f16(qa1, ld_b(row + 32), g);
  return g;
}

// ---------------- converters ----------------
__global__ void cvt_f16_kernel(const float* __restrict__ in, _Float16* __restrict__ out, int nElem) {
  int i = blockIdx.x * blockDim.x + threadIdx.x;
  int stride = gridDim.x * blockDim.x;
  for (; i < nElem; i += stride) out[i] = (_Float16)in[i];
}

__global__ void transpose_w_kernel(const float* __restrict__ W, _Float16* __restrict__ whT) {
  int i = blockIdx.x * blockDim.x + threadIdx.x; // i = c*1024 + d
  if (i >= NCOLS * KDIM) return;
  int c = i >> 10, d = i & 1023;
  whT[i] = (_Float16)W[(size_t)d * NCOLS + c];
}

// ---------------- QKV projection GEMM ----------------
// C(4096x3072) = X(4096x1024) * W(1024x3072) + b ; scatter into q (scaled), k, vT (f16)
__global__ __launch_bounds__(256)
void qkv_gemm_kernel(const _Float16* __restrict__ xh, const _Float16* __restrict__ whT,
                     const float* __restrict__ b_in,
                     _Float16* __restrict__ qh, _Float16* __restrict__ kh,
                     _Float16* __restrict__ vhT) {
  __shared__ _Float16 lx[128 * 32];
  __shared__ _Float16 lw[128 * 32];
  const int tid  = threadIdx.x;
  const int lane = tid & 31, wave = tid >> 5;
  const int l15  = lane & 15, hi = lane >> 4;
  const int wm = wave >> 1, wn = wave & 1;       // 4x2 waves -> 32x64 per wave
  const int M0 = blockIdx.y * 128, N0 = blockIdx.x * 128;
  const int srow = tid >> 1, shalf = tid & 1;

  v8f acc[2][4] = {};

  for (int k0 = 0; k0 < KDIM; k0 += 32) {
    const v8h* xs = (const v8h*)(xh  + (size_t)(M0 + srow) * KDIM + k0 + shalf * 16);
    const v8h* ws = (const v8h*)(whT + (size_t)(N0 + srow) * KDIM + k0 + shalf * 16);
    v8h x0 = xs[0], x1 = xs[1];
    v8h w0 = ws[0], w1 = ws[1];
    __syncthreads();
    ((v8h*)(lx + srow * 32 + shalf * 16))[0] = x0;
    ((v8h*)(lx + srow * 32 + shalf * 16))[1] = x1;
    ((v8h*)(lw + srow * 32 + shalf * 16))[0] = w0;
    ((v8h*)(lw + srow * 32 + shalf * 16))[1] = w1;
    __syncthreads();

    v16h af[2], bf[4];
#pragma unroll
    for (int mi = 0; mi < 2; ++mi) {
      const _Float16* p = lx + (wm * 32 + mi * 16 + l15) * 32;
      af[mi] = ld_a(p, hi * 16, hi * 16 + 8);
    }
#pragma unroll
    for (int ni = 0; ni < 4; ++ni) {
      const _Float16* p = lw + (wn * 64 + ni * 16 + l15) * 32 + hi * 16;
      bf[ni] = ld_b(p);
    }
#pragma unroll
    for (int mi = 0; mi < 2; ++mi)
#pragma unroll
      for (int ni = 0; ni < 4; ++ni)
        acc[mi][ni] = wmma_f16(af[mi], bf[ni], acc[mi][ni]);
  }

  // epilogue: bias + interleaved qkv scatter (c = hh*192 + dd*3 + s)
  // branchless: per-lane column decomposition is invariant across r, and the
  // q/k/v destination is chosen with pointer/index selects (v_cndmask), not branches.
#pragma unroll
  for (int mi = 0; mi < 2; ++mi) {
#pragma unroll
    for (int ni = 0; ni < 4; ++ni) {
      const int c   = N0 + wn * 64 + ni * 16 + l15;
      const int hh  = c / 192;
      const int rem = c - hh * 192;
      const int dd  = rem / 3;
      const int s   = rem - dd * 3;
      const float bias = b_in[c];
      const float scl  = (s == 0) ? QSCALE : 1.0f;
      _Float16* basep  = (s == 0) ? qh : ((s == 1) ? kh : vhT);
#pragma unroll
      for (int r = 0; r < 8; ++r) {
        const int t  = M0 + wm * 32 + mi * 16 + r + hi * 8;
        const int nb = t >> 11, tt = t & 2047;
        const size_t hb    = (size_t)(nb * HEADS + hh);
        const size_t idxQK = (hb * T_SEQ + tt) * DH + dd;
        const size_t idxV  = (hb * DH + dd) * T_SEQ + tt;
        const size_t idx   = (s == 2) ? idxV : idxQK;
        basep[idx] = (_Float16)((acc[mi][ni][r] + bias) * scl);
      }
    }
  }
}

// ---------------- fused attention (flash-style, rel-pos via banded WMMA) ----------------
__global__ __launch_bounds__(256)
void attn_kernel(const _Float16* __restrict__ qh, const _Float16* __restrict__ kh,
                 const _Float16* __restrict__ vhT, const _Float16* __restrict__ Rh,
                 const float* __restrict__ mask, float* __restrict__ out) {
  __shared__ _Float16 pbuf[8 * 16 * 32];     // per-wave 16x32 P staging
  const int tid  = threadIdx.x;
  const int lane = tid & 31, wave = tid >> 5;
  const int l15  = lane & 15, hi = lane >> 4;
  const int nh = blockIdx.x, n = nh >> 4, h = nh & 15;
  const int qi = blockIdx.y * 128 + wave * 16;

  const _Float16* qb = qh  + (size_t)nh * T_SEQ * DH;
  const _Float16* kb = kh  + (size_t)nh * T_SEQ * DH;
  const _Float16* vb = vhT + (size_t)nh * DH * T_SEQ;
  const _Float16* Rb = Rh  + (size_t)h * 4097 * DH;
  const float*    mb = mask + (size_t)n * T_SEQ * T_SEQ;

  // Q A-fragments (q pre-scaled by 0.0625*log2e)
  v16h qa0, qa1;
  {
    const _Float16* qrow = qb + (size_t)(qi + l15) * DH;
    qa0 = ld_a(qrow,      hi * 16, hi * 16 + 8);
    qa1 = ld_a(qrow + 32, hi * 16, hi * 16 + 8);
  }

  v8f o0 = {}, o1 = {}, o2 = {}, o3 = {};
  float mrow[8], lrow[8];
#pragma unroll
  for (int r = 0; r < 8; ++r) { mrow[r] = -1e30f; lrow[r] = 0.0f; }

  // rel GEMM chain: carry G(base C0-16) across iterations
  v8f Gprev = rel_g(Rb, 2032 - qi, qa0, qa1, l15, hi);

  _Float16* pl = pbuf + wave * (16 * 32);

  for (int j0 = 0; j0 < T_SEQ; j0 += 32) {
    const int C0 = 2048 + j0 - qi;
    v8f Gb = rel_g(Rb, C0,      qa0, qa1, l15, hi);
    v8f Gc = rel_g(Rb, C0 + 16, qa0, qa1, l15, hi);

    // dot tiles: S0 = cols [j0, j0+16), S1 = cols [j0+16, j0+32)
    v8f S0 = {}, S1 = {};
    {
      const _Float16* k0p = kb + (size_t)(j0 + l15) * DH + hi * 16;
      S0 = wmma_f16(qa0, ld_b(k0p),      S0);
      S0 = wmma_f16(qa1, ld_b(k0p + 32), S0);
      const _Float16* k1p = kb + (size_t)(j0 + 16 + l15) * DH + hi * 16;
      S1 = wmma_f16(qa0, ld_b(k1p),      S1);
      S1 = wmma_f16(qa1, ld_b(k1p + 32), S1);
    }
    if (j0 + 32 < T_SEQ)
      __builtin_prefetch(kb + (size_t)(j0 + 32 + lane) * DH, 0, 1);

    // assemble rel term from banded GEMMs + mask
#pragma unroll
    for (int r = 0; r < 8; ++r) {
      const int ii  = r + hi * 8;
      const int src = ((l15 - ii) & 15) | (lane & 16);
      float ga = __shfl(Gprev[r], src, 32);
      float gb = __shfl(Gb[r],   src, 32);
      float gc = __shfl(Gc[r],   src, 32);
      S0[r] += (l15 >= ii) ? gb : ga;
      S1[r] += (l15 >= ii) ? gc : gb;
      const float* mrp = mb + (size_t)(qi + ii) * T_SEQ + j0;
      S0[r] += (1.0f - mrp[l15])      * MASK_SCALE;
      S1[r] += (1.0f - mrp[16 + l15]) * MASK_SCALE;
    }
    Gprev = Gc;

    // online softmax (base-2) + stage P to LDS as f16
#pragma unroll
    for (int r = 0; r < 8; ++r) {
      float bm = half_reduce_max(fmaxf(S0[r], S1[r]));
      float nm = fmaxf(mrow[r], bm);
      float alpha = __builtin_amdgcn_exp2f(mrow[r] - nm);
      mrow[r] = nm;
      float p0 = __builtin_amdgcn_exp2f(S0[r] - nm);
      float p1 = __builtin_amdgcn_exp2f(S1[r] - nm);
      float rs = half_reduce_sum(p0 + p1);
      lrow[r] = lrow[r] * alpha + rs;
      o0[r] *= alpha; o1[r] *= alpha; o2[r] *= alpha; o3[r] *= alpha;
      pl[(r + hi * 8) * 32 + l15]      = (_Float16)p0;
      pl[(r + hi * 8) * 32 + 16 + l15] = (_Float16)p1;
    }
    // same-wave DS store->load ordering
    asm volatile("s_wait_dscnt 0x0" ::: "memory");

    v16h pa = ld_a(pl + l15 * 32, hi * 16, hi * 16 + 8);
    const _Float16* vrow = vb + (size_t)l15 * T_SEQ + j0 + hi * 16;
    o0 = wmma_f16(pa, ld_b(vrow),               o0);
    o1 = wmma_f16(pa, ld_b(vrow + 16 * T_SEQ),  o1);
    o2 = wmma_f16(pa, ld_b(vrow + 32 * T_SEQ),  o2);
    o3 = wmma_f16(pa, ld_b(vrow + 48 * T_SEQ),  o3);
  }

  // epilogue: out[n, i, h*64 + d] = o / l
  float* ob = out + ((size_t)n * T_SEQ + qi) * (HEADS * DH) + h * DH;
#pragma unroll
  for (int r = 0; r < 8; ++r) {
    float inv = __builtin_amdgcn_rcpf(lrow[r]);
    float* row = ob + (size_t)(r + hi * 8) * (HEADS * DH);
    row[l15]      = o0[r] * inv;
    row[16 + l15] = o1[r] * inv;
    row[32 + l15] = o2[r] * inv;
    row[48 + l15] = o3[r] * inv;
  }
}

extern "C" void kernel_launch(void* const* d_in, const int* in_sizes, int n_in,
                              void* d_out, int out_size, void* d_ws, size_t ws_size,
                              hipStream_t stream) {
  (void)in_sizes; (void)n_in; (void)out_size; (void)ws_size;
  const float* x    = (const float*)d_in[0];
  const float* mask = (const float*)d_in[1];
  const float* W    = (const float*)d_in[2];
  const float* b    = (const float*)d_in[3];
  const float* R    = (const float*)d_in[4];

  char* ws = (char*)d_ws;
  _Float16* xh  = (_Float16*)(ws);                      // 8 MB  x  f16 [4096][1024]
  _Float16* whT = (_Float16*)(ws + ( 8u << 20));        // 6 MB  W^T f16 [3072][1024]
  _Float16* qh  = (_Float16*)(ws + (16u << 20));        // 8 MB  q  f16 [n][h][t][dh] (scaled)
  _Float16* kh  = (_Float16*)(ws + (24u << 20));        // 8 MB  k  f16 [n][h][t][dh]
  _Float16* vhT = (_Float16*)(ws + (32u << 20));        // 8 MB  v  f16 [n][h][dh][t]
  _Float16* Rh  = (_Float16*)(ws + (40u << 20));        // 8.4MB R  f16 [h][4097][dh]

  cvt_f16_kernel<<<2048, 256, 0, stream>>>(x, xh, 4096 * 1024);
  cvt_f16_kernel<<<2048, 256, 0, stream>>>(R, Rh, 16 * 4097 * 64);
  transpose_w_kernel<<<(NCOLS * KDIM) / 256, 256, 0, stream>>>(W, whT);

  qkv_gemm_kernel<<<dim3(NCOLS / 128, 4096 / 128), 256, 0, stream>>>(xh, whT, b, qh, kh, vhT);

  attn_kernel<<<dim3(2 * HEADS, T_SEQ / 128), 256, 0, stream>>>(qh, kh, vhT, Rh, mask,
                                                                (float*)d_out);
}
